// Decoder_Cond_55576876810547
// MI455X (gfx1250) — matrix operand order repaired and checked by reading, not compiled, not run on hardware
//
#include <hip/hip_runtime.h>

// ---------------------------------------------------------------------------
// CDNA5 (gfx1250, wave32) LSTM decoder. bf16 WMMA (v_wmma_f32_16x16x32_bf16)
// for all GEMMs; LSTM cell + bias fused into the GEMM; embedding gather fused
// into the layer-0 GEMM A-fragment loads; h-state ping-pong buffered.
// K loop split into branch-free per-segment loops (segment choice hoisted).
// ---------------------------------------------------------------------------

typedef __attribute__((ext_vector_type(16))) __bf16        v16bf;
typedef __attribute__((ext_vector_type(8)))  float         v8f;
typedef __attribute__((ext_vector_type(4)))  unsigned int  u32x4;

union BF16Tile { v16bf v; u32x4 q[2]; };

#define B_   32
#define T_   128
#define V_   16000
#define E_   512
#define Z_   128
#define H_   1024
#define K0_  1664         // E + Z + H   (layer0 concatenated K)
#define K1_  2048         // H + H       (layer1 concatenated K)
#define START_ID_ 1

__device__ __forceinline__ unsigned short f2bf(float f) {
  unsigned int u = __float_as_uint(f);
  u += 0x7fffu + ((u >> 16) & 1u);      // round-to-nearest-even
  return (unsigned short)(u >> 16);
}
__device__ __forceinline__ float sigm(float x) { return 1.0f / (1.0f + __expf(-x)); }

// one branch-free GEMM segment: acc[g] += A(seg) x B(Wt rows, 4 gate blocks)
__device__ __forceinline__ void gemm_seg(v8f* acc, const unsigned short* __restrict__ src,
                                         const unsigned short* __restrict__ wb,
                                         size_t gstep, int klen) {
  for (int k0 = 0; k0 < klen; k0 += 32) {
    BF16Tile a;
    a.q[0] = *(const u32x4*)(src + k0);         // K = k0 + khalf*8 .. +8
    a.q[1] = *(const u32x4*)(src + k0 + 16);    // K = k0+16 + khalf*8 .. +8
    __builtin_prefetch((const void*)(wb + k0 + 32), 0, 0);
#pragma unroll
    for (int g = 0; g < 4; ++g) {
      BF16Tile bt;
      const u32x4* bp = (const u32x4*)(wb + k0 + g * gstep);
      bt.q[0] = bp[0];
      bt.q[1] = bp[2];
      acc[g] = __builtin_amdgcn_wmma_f32_16x16x32_bf16(
          false, a.v, false, bt.v, (short)0, acc[g], false, false);
    }
  }
}

// ---------------------------- prep kernels ---------------------------------

__global__ void kconv_f32_bf16(const float* __restrict__ in,
                               unsigned short* __restrict__ out, int n) {
  int i = blockIdx.x * blockDim.x + threadIdx.x;
  if (i < n) out[i] = f2bf(in[i]);
}

// in is [K,N] row-major f32; out[n*ldo + koff + k] = bf16(in[k*N+n])
__global__ void ktr_f32_bf16(const float* __restrict__ in,
                             unsigned short* __restrict__ out,
                             int K, int N, int koff, int ldo) {
  long long i = (long long)blockIdx.x * blockDim.x + threadIdx.x;
  long long total = (long long)K * N;
  if (i >= total) return;
  int k = (int)(i / N), n = (int)(i % N);
  out[(long long)n * ldo + koff + k] = f2bf(in[i]);
}

__global__ void kzero_f32(float* p, int n) {
  int i = blockIdx.x * blockDim.x + threadIdx.x;
  if (i < n) p[i] = 0.0f;
}
__global__ void kzero_u16(unsigned short* p, int n) {
  int i = blockIdx.x * blockDim.x + threadIdx.x;
  if (i < n) p[i] = 0;
}
__global__ void kset_tok(int* tok) {
  if (threadIdx.x < B_) tok[threadIdx.x] = START_ID_;
}

// --------------------------- per-step kernels ------------------------------

// gates = A[32,K] @ Wt^T  (Wt stored [4H, K] bf16), bias preloaded into the
// accumulator, LSTM cell fused in the epilogue. A-fragments gathered on the
// fly per segment:
//   LAYER 0:  [emb_bf[tok(row)] (512) | z_bf (128) | hA = h0_prev (1024)]
//   LAYER 1:  [hA = h0_cur (1024)     | hB = h1_prev (1024)]
// Segment loops are branch-free; all boundaries 32-aligned; every fragment
// load is a contiguous 16B b128.
// grid = 16 blocks x 256 thr = 128 waves = 2 Mtiles * 64 Htiles.
template <int LAYER>
__global__ void klstm_step(const int* __restrict__ x, const int* __restrict__ prev_tok,
                           int t, int fed,
                           const unsigned short* __restrict__ emb_bf,
                           const unsigned short* __restrict__ z_bf,
                           const unsigned short* __restrict__ hA,
                           const unsigned short* __restrict__ hB,
                           const unsigned short* __restrict__ Wt,
                           const float* __restrict__ bias,
                           float* __restrict__ c,
                           unsigned short* __restrict__ hout) {
  const int K = (LAYER == 0) ? K0_ : K1_;
  int wave  = (blockIdx.x * blockDim.x + threadIdx.x) >> 5;
  int lane  = threadIdx.x & 31;
  int m0    = (wave & 1) * 16;
  int n0    = (wave >> 1) * 16;
  int nl    = lane & 15;
  int khalf = lane >> 4;
  int row   = m0 + nl;           // this lane's A (batch) row
  int hcol  = n0 + nl;           // this lane's C/D (hidden) column

  // preload bias into the accumulators (C value depends only on hcol)
  v8f acc[4];
#pragma unroll
  for (int g = 0; g < 4; ++g) {
    float bv = bias[g * H_ + hcol];
#pragma unroll
    for (int r = 0; r < 8; ++r) acc[g][r] = bv;
  }

  const unsigned short* wb = Wt + (size_t)(n0 + nl) * K + khalf * 8;
  const size_t gstep = (size_t)H_ * K;
  const unsigned short* hArow = hA + row * H_ + khalf * 8;

  if constexpr (LAYER == 0) {
    int tok = fed ? ((t == 0) ? START_ID_ : x[row * T_ + (t - 1)]) : prev_tok[row];
    const unsigned short* embrow = emb_bf + (size_t)tok * E_ + khalf * 8;
    const unsigned short* zrow   = z_bf + row * Z_ + khalf * 8;
    gemm_seg(acc, embrow, wb,           gstep, E_);
    gemm_seg(acc, zrow,   wb + E_,      gstep, Z_);
    gemm_seg(acc, hArow,  wb + E_ + Z_, gstep, H_);
  } else {
    const unsigned short* hBrow = hB + row * H_ + khalf * 8;
    gemm_seg(acc, hArow, wb,      gstep, H_);
    gemm_seg(acc, hBrow, wb + H_, gstep, H_);
  }

#pragma unroll
  for (int r = 0; r < 8; ++r) {
    int   br = m0 + khalf * 8 + r;              // C/D layout: lanes16-31 -> M+8
    float iv = sigm(acc[0][r]);
    float fv = sigm(acc[1][r]);
    float gv = tanhf(acc[2][r]);
    float ov = sigm(acc[3][r]);
    float cp = c[(size_t)br * H_ + hcol];
    float cn = fv * cp + iv * gv;
    float hv = ov * tanhf(cn);
    c[(size_t)br * H_ + hcol] = cn;
    hout[(size_t)br * H_ + hcol] = f2bf(hv);
  }
}

// logits[32,16000] = h1[32,1024] @ Wo^T (Wo stored [16000,1024] bf16) + bout
// grid = 250 blocks x 256 thr = 2000 waves = 2 Mtiles * 1000 Ntiles.
__global__ void klogits(const unsigned short* __restrict__ act,
                        const unsigned short* __restrict__ Wt,
                        const float* __restrict__ bias,
                        float* __restrict__ out) {
  const int K = H_;
  int wave  = (blockIdx.x * blockDim.x + threadIdx.x) >> 5;
  int lane  = threadIdx.x & 31;
  int m0    = (wave % 2) * 16;
  int n0    = (wave / 2) * 16;
  int nl    = lane & 15;
  int khalf = lane >> 4;
  int col   = n0 + nl;

  v8f acc;
  {
    float bv = bias[col];
#pragma unroll
    for (int r = 0; r < 8; ++r) acc[r] = bv;
  }
  const unsigned short* arow = act + (size_t)(m0 + nl) * K + khalf * 8;
  const unsigned short* wb   = Wt  + (size_t)col * K + khalf * 8;

  for (int k0 = 0; k0 < K; k0 += 32) {
    BF16Tile a, bt;
    a.q[0] = *(const u32x4*)(arow + k0);
    a.q[1] = *(const u32x4*)(arow + k0 + 16);
    const u32x4* bp = (const u32x4*)(wb + k0);
    bt.q[0] = bp[0];
    bt.q[1] = bp[2];
    acc = __builtin_amdgcn_wmma_f32_16x16x32_bf16(
        false, a.v, false, bt.v, (short)0, acc, false, false);
  }
#pragma unroll
  for (int r = 0; r < 8; ++r) {
    int br = m0 + khalf * 8 + r;
    out[(size_t)br * V_ + col] = acc[r];
  }
}

// softmax over V=16000, write p to d_out, gumbel(hash)-argmax sample token.
__global__ void ksample(const float* __restrict__ logits, float* __restrict__ dout,
                        int* __restrict__ prev_tok, int t, int stream_id,
                        long long s_base, long long p_base) {
  int b = blockIdx.x, tid = threadIdx.x;
  __shared__ float sred[256];
  __shared__ int   sidx[256];
  const float* lr = logits + (size_t)b * V_;

  float mx = -3.4e38f;
  for (int v = tid; v < V_; v += 256) mx = fmaxf(mx, lr[v]);
  sred[tid] = mx; __syncthreads();
  for (int s = 128; s > 0; s >>= 1) {
    if (tid < s) sred[tid] = fmaxf(sred[tid], sred[tid + s]);
    __syncthreads();
  }
  mx = sred[0]; __syncthreads();

  float sum = 0.0f;
  for (int v = tid; v < V_; v += 256) sum += __expf(lr[v] - mx);
  sred[tid] = sum; __syncthreads();
  for (int s = 128; s > 0; s >>= 1) {
    if (tid < s) sred[tid] += sred[tid + s];
    __syncthreads();
  }
  sum = sred[0]; __syncthreads();
  float logZ = mx + __logf(sum);

  float best = -3.4e38f; int bestv = 0;
  float* pout = dout + p_base + ((size_t)b * T_ + t) * V_;
  for (int v = tid; v < V_; v += 256) {
    float lp = lr[v] - logZ;
    pout[v] = __expf(lp);
    unsigned int h = (unsigned int)v * 2654435761u +
                     (unsigned int)(b * 131 + t * 7919 + stream_id * 104729 + 12345);
    h ^= h >> 16; h *= 0x85ebca6bu; h ^= h >> 13; h *= 0xc2b2ae35u; h ^= h >> 16;
    float u = (float)(h & 0x00FFFFFFu) * (1.0f / 16777216.0f) + 5.96e-8f;
    float g = -__logf(-__logf(u));
    float s = lp + g;
    if (s > best) { best = s; bestv = v; }
  }
  sred[tid] = best; sidx[tid] = bestv; __syncthreads();
  for (int s = 128; s > 0; s >>= 1) {
    if (tid < s && sred[tid + s] > sred[tid]) {
      sred[tid] = sred[tid + s]; sidx[tid] = sidx[tid + s];
    }
    __syncthreads();
  }
  if (tid == 0) {
    prev_tok[b] = sidx[0];
    dout[s_base + (size_t)b * T_ + t] = (float)sidx[0];
  }
}

// ------------------------------- host --------------------------------------

extern "C" void kernel_launch(void* const* d_in, const int* in_sizes, int n_in,
                              void* d_out, int out_size, void* d_ws, size_t ws_size,
                              hipStream_t stream) {
  (void)in_sizes; (void)n_in; (void)out_size; (void)ws_size;
  const int*   x    = (const int*)  d_in[0];
  const float* z    = (const float*)d_in[1];
  const float* emb  = (const float*)d_in[2];
  const float* Wx0  = (const float*)d_in[3];
  const float* Wh0  = (const float*)d_in[4];
  const float* b0   = (const float*)d_in[5];
  const float* Wx1  = (const float*)d_in[6];
  const float* Wh1  = (const float*)d_in[7];
  const float* b1   = (const float*)d_in[8];
  const float* Wout = (const float*)d_in[9];
  const float* bout = (const float*)d_in[10];
  float* outf = (float*)d_out;

  // workspace layout (256B aligned slabs)
  char* base = (char*)d_ws;
  size_t off = 0;
  auto alloc = [&](size_t bytes) -> char* {
    off = (off + 255) & ~(size_t)255;
    char* p = base + off; off += bytes; return p;
  };
  const size_t VE = (size_t)V_ * E_;
  unsigned short* emb_bf = (unsigned short*)alloc(VE * 2);
  unsigned short* z_bf   = (unsigned short*)alloc((size_t)B_ * Z_ * 2);
  unsigned short* W0     = (unsigned short*)alloc((size_t)4 * H_ * K0_ * 2);
  unsigned short* W1     = (unsigned short*)alloc((size_t)4 * H_ * K1_ * 2);
  unsigned short* Wo     = (unsigned short*)alloc((size_t)V_ * H_ * 2);
  unsigned short* h0b[2], *h1b[2];
  h0b[0] = (unsigned short*)alloc((size_t)B_ * H_ * 2);
  h0b[1] = (unsigned short*)alloc((size_t)B_ * H_ * 2);
  h1b[0] = (unsigned short*)alloc((size_t)B_ * H_ * 2);
  h1b[1] = (unsigned short*)alloc((size_t)B_ * H_ * 2);
  float* c0     = (float*)alloc((size_t)B_ * H_ * 4);
  float* c1     = (float*)alloc((size_t)B_ * H_ * 4);
  float* logits = (float*)alloc((size_t)B_ * V_ * 4);
  int*   ptok   = (int*)  alloc((size_t)B_ * 4);

  // ---- weight conversion (f32 -> bf16, GEMM-B pre-transposed to [N,K]) ----
  kconv_f32_bf16<<<(int)((VE + 255) / 256), 256, 0, stream>>>(emb, emb_bf, (int)VE);
  kconv_f32_bf16<<<(B_ * Z_ + 255) / 256, 256, 0, stream>>>(z, z_bf, B_ * Z_);
  ktr_f32_bf16<<<(int)(((size_t)(E_ + Z_) * 4 * H_ + 255) / 256), 256, 0, stream>>>(Wx0,  W0, E_ + Z_, 4 * H_, 0,       K0_);
  ktr_f32_bf16<<<(int)(((size_t)H_ * 4 * H_ + 255) / 256), 256, 0, stream>>>(Wh0,  W0, H_, 4 * H_, E_ + Z_, K0_);
  ktr_f32_bf16<<<(int)(((size_t)H_ * 4 * H_ + 255) / 256), 256, 0, stream>>>(Wx1,  W1, H_, 4 * H_, 0,       K1_);
  ktr_f32_bf16<<<(int)(((size_t)H_ * 4 * H_ + 255) / 256), 256, 0, stream>>>(Wh1,  W1, H_, 4 * H_, H_,      K1_);
  ktr_f32_bf16<<<(int)(((size_t)H_ * V_ + 255) / 256), 256, 0, stream>>>(Wout, Wo, H_, V_,     0,       H_);

  const long long BT  = (long long)B_ * T_;
  const long long BTV = BT * V_;

  for (int pass = 0; pass < 2; ++pass) {         // 0 = free-running, 1 = fed
    long long s_base = (pass == 0) ? 0 : (BT + BTV);
    long long p_base = (pass == 0) ? BT : (2 * BT + BTV);
    int fed = pass;                              // stream_id matches pass

    kzero_f32<<<(B_ * H_ + 255) / 256, 256, 0, stream>>>(c0, B_ * H_);
    kzero_f32<<<(B_ * H_ + 255) / 256, 256, 0, stream>>>(c1, B_ * H_);
    kzero_u16<<<(B_ * H_ + 255) / 256, 256, 0, stream>>>(h0b[0], B_ * H_);
    kzero_u16<<<(B_ * H_ + 255) / 256, 256, 0, stream>>>(h1b[0], B_ * H_);
    kset_tok<<<1, 32, 0, stream>>>(ptok);

    for (int t = 0; t < T_; ++t) {
      unsigned short* h0_prev = h0b[t & 1];
      unsigned short* h0_cur  = h0b[(t + 1) & 1];
      unsigned short* h1_prev = h1b[t & 1];
      unsigned short* h1_cur  = h1b[(t + 1) & 1];

      // layer0: A = [emb[tok], z, h0_prev] -> writes h0_cur
      klstm_step<0><<<16, 256, 0, stream>>>(
          x, ptok, t, fed, emb_bf, z_bf, h0_prev, (unsigned short*)nullptr,
          W0, b0, c0, h0_cur);
      // layer1: A = [h0_cur, h1_prev] -> writes h1_cur
      klstm_step<1><<<16, 256, 0, stream>>>(
          x, ptok, t, fed, emb_bf, z_bf, h0_cur, h1_prev,
          W1, b1, c1, h1_cur);
      klogits<<<250, 256, 0, stream>>>(h1_cur, Wo, bout, logits);
      ksample<<<B_, 256, 0, stream>>>(logits, outf, ptok, t, pass, s_base, p_base);
    }
  }
}
